// CausalSelfAttention_7980049236635
// MI455X (gfx1250) — matrix operand order repaired and checked by reading, not compiled, not run on hardware
//
#include <hip/hip_runtime.h>

// ---------------------------------------------------------------------------
// CausalSelfAttention for MI455X (gfx1250): B=4, T=2048, C=1024, H=16, HD=64
// v_wmma_f32_16x16x32_f16 everywhere; global->LDS staging via
// global_load_async_to_lds_b128 (ASYNCcnt) with double-buffered LDS.
// ---------------------------------------------------------------------------

typedef __attribute__((ext_vector_type(16))) _Float16 v16h;
typedef __attribute__((ext_vector_type(8)))  _Float16 v8h;
typedef __attribute__((ext_vector_type(8)))  float    v8f;

#define NB   4
#define NT   2048
#define NC   1024
#define NH   16
#define NHD  64
#define N3C  3072

union F16x16 { v16h v; v8h h[2]; };

// Async 16-byte global->LDS copy (CDNA5 GLOBAL_LOAD_ASYNC_TO_LDS_B128).
// LDS flat address low 32 bits are the LDS byte address (aperture in [63:32]).
__device__ __forceinline__ void async_copy_b128(void* lds_dst, const void* gsrc) {
    unsigned lds = (unsigned)(size_t)lds_dst;
    asm volatile("global_load_async_to_lds_b128 %0, %1, off"
                 :: "v"(lds), "v"(gsrc) : "memory");
}
__device__ __forceinline__ void wait_async() {
    asm volatile("s_wait_asynccnt 0x0" ::: "memory");
}

// A-fragment (16x32 f16) from row-major [M][K-span] LDS tile.
// lanes 0-15 -> M=0..15; halves 0..7 = K klo..+7, halves 8..15 = K klo+16..+23,
// klo = (lane<16 ? 0 : 8).
__device__ __forceinline__ v16h load_frag_a(const _Float16* base, int stride, int lane) {
    int m   = lane & 15;
    int klo = (lane < 16) ? 0 : 8;
    const _Float16* p = base + (size_t)m * stride + klo;
    F16x16 r;
    r.h[0] = *(const v8h*)(p);
    r.h[1] = *(const v8h*)(p + 16);
    return r.v;
}

// B-fragment (32x16 f16) from an [N][K] (transposed) LDS tile.
// lane -> N = lane%16; halves 0..7 = K klo..+7, halves 8..15 = K klo+16..+23,
// klo = (lane<16 ? 0 : 8).
__device__ __forceinline__ v16h load_frag_b(const _Float16* base, int stride, int lane) {
    int n   = lane & 15;
    int klo = (lane < 16) ? 0 : 8;
    const _Float16* p = base + (size_t)n * stride + klo;
    F16x16 r;
    r.h[0] = *(const v8h*)(p);
    r.h[1] = *(const v8h*)(p + 16);
    return r.v;
}

// ---------------------------------------------------------------------------
// Kernel 0: fp32 -> fp16 conversion (grid-stride).
// ---------------------------------------------------------------------------
__global__ void cvt_f32_to_f16(const float* __restrict__ src,
                               _Float16* __restrict__ dst, int n) {
    int i = blockIdx.x * blockDim.x + threadIdx.x;
    int stride = gridDim.x * blockDim.x;
    for (; i < n; i += stride) dst[i] = (_Float16)src[i];
}

// ---------------------------------------------------------------------------
// WMMA GEMM: C[M,N] = A[M,K] * B[K,N], f16 in, f32 accum.
// 256 threads = 8 waves; block tile 128x128, K-step 64 (2 WMMA chunks),
// double-buffered LDS, async A staging.
// MODE 0: scatter into q/k/v [B,H,T,HD] f16 (scale 0.125 on q).
// MODE 1: fp32 epilogue to out[M,N].
// ---------------------------------------------------------------------------
#define BM 128
#define BN 128
#define BK 64

template <int MODE>
__global__ __launch_bounds__(256)
void gemm_wmma(const _Float16* __restrict__ A, const _Float16* __restrict__ Bg,
               _Float16* __restrict__ o0, _Float16* __restrict__ o1,
               _Float16* __restrict__ o2, float* __restrict__ fo,
               int Mdim, int Ndim, int Kdim) {
    __shared__ _Float16 As[2][BM][BK];
    __shared__ _Float16 Bs[2][BN][BK];   // transposed: [N][K]

    const int tid    = threadIdx.x;
    const int lane   = tid & 31;
    const int wid    = tid >> 5;
    const int wave_m = wid >> 2;      // 0..1
    const int wave_n = wid & 3;       // 0..3
    const int bm0    = blockIdx.y * BM;
    const int bn0    = blockIdx.x * BN;

    v8f acc[4][2];
#pragma unroll
    for (int i = 0; i < 4; ++i)
#pragma unroll
        for (int j = 0; j < 2; ++j) acc[i][j] = (v8f)0.0f;

    const int nk = Kdim / BK;

    auto stage = [&](int kt, int buf) {
        const int k0 = kt * BK;
        // A tile: 128 rows x 64 halves, async 16B chunks (4 per thread).
#pragma unroll
        for (int c = tid; c < (BM * BK) / 8; c += 256) {
            int row  = c >> 3;                 // BK/8 = 8 chunks per row
            int colh = (c & 7) * 8;
            async_copy_b128(&As[buf][row][colh],
                            &A[(size_t)(bm0 + row) * Kdim + k0 + colh]);
        }
        // B tile transposed: read 8 contiguous N, scatter into [N][K].
#pragma unroll
        for (int c = tid; c < (BN * BK) / 8; c += 256) {
            int kk = c >> 4;                   // BN/8 = 16 chunks per k-row
            int nn = (c & 15) * 8;
            v8h val = *(const v8h*)&Bg[(size_t)(k0 + kk) * Ndim + bn0 + nn];
#pragma unroll
            for (int j = 0; j < 8; ++j) Bs[buf][nn + j][kk] = val[j];
        }
    };

    stage(0, 0);
    for (int kt = 0; kt < nk; ++kt) {
        const int buf = kt & 1;
        wait_async();
        __syncthreads();
        if (kt + 1 < nk) stage(kt + 1, buf ^ 1);

#pragma unroll
        for (int c = 0; c < BK / 32; ++c) {
            v16h afr[4], bfr[2];
#pragma unroll
            for (int i = 0; i < 4; ++i)
                afr[i] = load_frag_a(&As[buf][wave_m * 64 + i * 16][c * 32],
                                     BK, lane);
#pragma unroll
            for (int j = 0; j < 2; ++j)
                bfr[j] = load_frag_b(&Bs[buf][wave_n * 32 + j * 16][c * 32],
                                     BK, lane);
#pragma unroll
            for (int i = 0; i < 4; ++i)
#pragma unroll
                for (int j = 0; j < 2; ++j)
                    acc[i][j] = __builtin_amdgcn_wmma_f32_16x16x32_f16(
                        false, afr[i], false, bfr[j], (short)0, acc[i][j],
                        false, false);
        }
    }

    // Epilogue. C layout: VGPR r, lane L -> M = r + (L>=16?8:0), N = L%16.
    const int hi8 = (lane >= 16) ? 8 : 0;
#pragma unroll
    for (int i = 0; i < 4; ++i) {
#pragma unroll
        for (int j = 0; j < 2; ++j) {
            int gm0 = bm0 + wave_m * 64 + i * 16;
            int gn0 = bn0 + wave_n * 32 + j * 16;
#pragma unroll
            for (int r = 0; r < 8; ++r) {
                int gm = gm0 + r + hi8;
                int gn = gn0 + (lane & 15);
                float val = acc[i][j][r];
                if (MODE == 0) {
                    int part = gn >> 10;       // 0:q 1:k 2:v
                    int cc   = gn & 1023;
                    int hh   = cc >> 6;
                    int dd   = cc & 63;
                    int bb   = gm >> 11;       // / T
                    int tt   = gm & 2047;
                    _Float16* dst = (part == 0) ? o0 : ((part == 1) ? o1 : o2);
                    float sv = (part == 0) ? val * 0.125f : val;
                    dst[(((size_t)bb * NH + hh) * NT + tt) * NHD + dd] =
                        (_Float16)sv;
                } else {
                    fo[(size_t)gm * Ndim + gn] = val;
                }
            }
        }
    }
}

// ---------------------------------------------------------------------------
// Kernel 2: flash attention. grid = (T/64, B*H), block = 128 (4 waves).
// Each wave owns 16 query rows; 64-key tiles double-buffered; online softmax.
// q pre-scaled by 1/sqrt(HD); mask replaces with -1e9 like the reference.
// ---------------------------------------------------------------------------
__global__ __launch_bounds__(128)
void attn_wmma(const _Float16* __restrict__ q, const _Float16* __restrict__ k,
               const _Float16* __restrict__ v, _Float16* __restrict__ y) {
    __shared__ _Float16 Qs[64][64];
    __shared__ _Float16 Ks[2][64][64];     // [key][hd] == [N][K] for QK^T
    __shared__ _Float16 Vt[2][64][64];     // [hd][key] == [N][K] for P*V
    __shared__ _Float16 Ps[4][16][64];     // per-wave P staging

    const int tid  = threadIdx.x;
    const int lane = tid & 31;
    const int w    = tid >> 5;             // wave id: q rows [16w,16w+16)
    const int bh   = blockIdx.y;           // 0..B*H-1
    const int qt0  = blockIdx.x * 64;
    const int hi8  = (lane >= 16) ? 8 : 0;

    const _Float16* qp = q + (size_t)bh * NT * NHD;
    const _Float16* kp = k + (size_t)bh * NT * NHD;
    const _Float16* vp = v + (size_t)bh * NT * NHD;

    auto stage = [&](int kt, int buf) {
        const int kt0 = kt * 64;
#pragma unroll
        for (int c = tid; c < (64 * 64) / 8; c += 128) {
            int row = c >> 3;
            int col = (c & 7) * 8;
            async_copy_b128(&Ks[buf][row][col],
                            &kp[(size_t)(kt0 + row) * NHD + col]);
            v8h vv = *(const v8h*)&vp[(size_t)(kt0 + row) * NHD + col];
#pragma unroll
            for (int j = 0; j < 8; ++j) Vt[buf][col + j][row] = vv[j];
        }
    };

    // Stage Q (async) and first K/V tile.
    for (int c = tid; c < (64 * 64) / 8; c += 128) {
        int row = c >> 3;
        int col = (c & 7) * 8;
        async_copy_b128(&Qs[row][col], &qp[(size_t)(qt0 + row) * NHD + col]);
    }
    stage(0, 0);

    v16h a_q[2];
    float m_state[8], l_state[8];
    v8f o_acc[4];
#pragma unroll
    for (int r = 0; r < 8; ++r) { m_state[r] = -1e30f; l_state[r] = 0.0f; }
#pragma unroll
    for (int n = 0; n < 4; ++n) o_acc[n] = (v8f)0.0f;

    const int ktiles = blockIdx.x + 1;     // causal: keys up to qt0+63
    for (int kt = 0; kt < ktiles; ++kt) {
        const int buf  = kt & 1;
        const int kt0  = kt * 64;
        const bool diag = (kt0 == qt0);

        wait_async();
        __syncthreads();
        if (kt + 1 < ktiles) stage(kt + 1, buf ^ 1);
        if (kt == 0) {
#pragma unroll
            for (int c = 0; c < 2; ++c)
                a_q[c] = load_frag_a(&Qs[w * 16][c * 32], 64, lane);
        }

        // Scores S = Q * K^T : 16 x 64 per wave.
        v8f s[4];
#pragma unroll
        for (int n = 0; n < 4; ++n) s[n] = (v8f)0.0f;
#pragma unroll
        for (int n = 0; n < 4; ++n)
#pragma unroll
            for (int c = 0; c < 2; ++c) {
                v16h bfr = load_frag_b(&Ks[buf][n * 16][c * 32], 64, lane);
                s[n] = __builtin_amdgcn_wmma_f32_16x16x32_f16(
                    false, a_q[c], false, bfr, (short)0, s[n], false, false);
            }

        // Causal mask on the diagonal tile (replace with -1e9).
        if (diag) {
#pragma unroll
            for (int n = 0; n < 4; ++n) {
                int kn = kt0 + n * 16 + (lane & 15);
#pragma unroll
                for (int r = 0; r < 8; ++r) {
                    int qm = qt0 + w * 16 + r + hi8;
                    if (kn > qm) s[n][r] = -1e9f;
                }
            }
        }

        // Online softmax per row (row = 16-lane half-wave group).
#pragma unroll
        for (int r = 0; r < 8; ++r) {
            float t = fmaxf(fmaxf(s[0][r], s[1][r]), fmaxf(s[2][r], s[3][r]));
#pragma unroll
            for (int msk = 1; msk < 16; msk <<= 1)
                t = fmaxf(t, __shfl_xor(t, msk, 32));
            float mn = fmaxf(m_state[r], t);
            float sc = __expf(m_state[r] - mn);
            m_state[r] = mn;
            float rs = 0.0f;
#pragma unroll
            for (int n = 0; n < 4; ++n) {
                float p = __expf(s[n][r] - mn);
                s[n][r] = p;
                rs += p;
            }
#pragma unroll
            for (int msk = 1; msk < 16; msk <<= 1)
                rs += __shfl_xor(rs, msk, 32);
            l_state[r] = l_state[r] * sc + rs;
#pragma unroll
            for (int n = 0; n < 4; ++n) o_acc[n][r] = o_acc[n][r] * sc;
        }

        // P: C-layout -> A-layout via per-wave LDS (same-wave DS is in-order).
#pragma unroll
        for (int n = 0; n < 4; ++n)
#pragma unroll
            for (int r = 0; r < 8; ++r)
                Ps[w][r + hi8][n * 16 + (lane & 15)] = (_Float16)s[n][r];

        // O += P * V.
#pragma unroll
        for (int c = 0; c < 2; ++c) {
            v16h a_p = load_frag_a(&Ps[w][0][c * 32], 64, lane);
#pragma unroll
            for (int n = 0; n < 4; ++n) {
                v16h bfr = load_frag_b(&Vt[buf][n * 16][c * 32], 64, lane);
                o_acc[n] = __builtin_amdgcn_wmma_f32_16x16x32_f16(
                    false, a_p, false, bfr, (short)0, o_acc[n], false, false);
            }
        }
    }

    // Epilogue: y[b, t, h*64 + d] = O / l  (f16 for the projection GEMM).
    const int bb = bh / NH;
    const int hh = bh % NH;
#pragma unroll
    for (int r = 0; r < 8; ++r) {
        float inv = 1.0f / l_state[r];
        int mr = w * 16 + r + hi8;
        int tt = qt0 + mr;
#pragma unroll
        for (int n = 0; n < 4; ++n) {
            int dd = n * 16 + (lane & 15);
            y[((size_t)bb * NT + tt) * NC + hh * NHD + dd] =
                (_Float16)(o_acc[n][r] * inv);
        }
    }
}

// ---------------------------------------------------------------------------
// Launch
// ---------------------------------------------------------------------------
extern "C" void kernel_launch(void* const* d_in, const int* in_sizes, int n_in,
                              void* d_out, int out_size, void* d_ws, size_t ws_size,
                              hipStream_t stream) {
    const float* x      = (const float*)d_in[0];   // (4,2048,1024)
    const float* W_attn = (const float*)d_in[1];   // (1024,3072)
    const float* W_proj = (const float*)d_in[2];   // (1024,1024)
    float* out          = (float*)d_out;           // (4,2048,1024)

    const size_t Ntok = (size_t)NB * NT;           // 8192
    char* ws = (char*)d_ws;
    size_t off = 0;
    auto take = [&](size_t elems) {
        _Float16* p = (_Float16*)(ws + off);
        off = (off + elems * sizeof(_Float16) + 255) & ~(size_t)255;
        return p;
    };
    _Float16* xh  = take(Ntok * NC);
    _Float16* wah = take((size_t)NC * N3C);
    _Float16* wph = take((size_t)NC * NC);
    _Float16* qb  = take((size_t)NB * NH * NT * NHD);
    _Float16* kb  = take((size_t)NB * NH * NT * NHD);
    _Float16* vb  = take((size_t)NB * NH * NT * NHD);
    _Float16* yh  = take(Ntok * NC);
    (void)ws_size;

    int nx = (int)(Ntok * NC), na = NC * N3C, np = NC * NC;
    cvt_f32_to_f16<<<dim3(4096), dim3(256), 0, stream>>>(x, xh, nx);
    cvt_f32_to_f16<<<dim3(4096), dim3(256), 0, stream>>>(W_attn, wah, na);
    cvt_f32_to_f16<<<dim3(2048), dim3(256), 0, stream>>>(W_proj, wph, np);

    // QKV: (8192 x 1024) * (1024 x 3072)
    gemm_wmma<0><<<dim3(N3C / BN, Ntok / BM), dim3(256), 0, stream>>>(
        xh, wah, qb, kb, vb, nullptr, (int)Ntok, N3C, NC);

    // Flash attention: grid (T/64, B*H)
    attn_wmma<<<dim3(NT / 64, NB * NH), dim3(128), 0, stream>>>(qb, kb, vb, yh);

    // Projection: (8192 x 1024) * (1024 x 1024) -> fp32 out
    gemm_wmma<1><<<dim3(NC / BN, Ntok / BM), dim3(256), 0, stream>>>(
        yh, wph, nullptr, nullptr, nullptr, out, (int)Ntok, NC, NC);
}
